// HA_NET_44392781971839
// MI455X (gfx1250) — compile-verified, matrix operand order checked by reading, not run on hardware
//
#include <hip/hip_runtime.h>
#include <math.h>

// ---------------------------------------------------------------------------
// HA-Net forward on gfx1250 (MI455X).  All GEMM-like math uses
// v_wmma_f32_16x16x32_f16.  B operands are pre-swizzled fragment-major so a
// lane's 8 fragment dwords are contiguous (2x b128 per fragment, coalesced);
// A operands are pair-packed row-major (2x b128 with immediate offsets).
// Register tiling: 4 N-tiles/wave (gemm), 7 (conv), 6 (recurrent) per A-frag.
//
// Workspace usage ~213 MB.
// ---------------------------------------------------------------------------

typedef __attribute__((ext_vector_type(16))) _Float16 v16h;
typedef __attribute__((ext_vector_type(8)))  _Float16 v8h;
typedef __attribute__((ext_vector_type(8)))  float    v8f;

union Frag { unsigned u[8]; v16h v; };

#define DEVFN static __device__ __forceinline__

// Pair index (dword) held by VGPR j for lane-half hi inside a 16-pair k-tile
// (ISA 7.12.2 16-bit layouts).
DEVFN int kpo(int j, int hi) { return ((j & 4) << 1) + (j & 3) + (hi << 2); }

DEVFN v8f wmma_f16(const Frag& a, const Frag& b, v8f c) {
  return __builtin_amdgcn_wmma_f32_16x16x32_f16(
      false, a.v, false, b.v, (short)0, c, false, false);
}

// A fragment: 8 dwords at p[0..3] and p[8..11]  (p already includes +4*hi)
DEVFN void loadA(Frag& a, const unsigned* __restrict__ p) {
#pragma unroll
  for (int j = 0; j < 4; ++j) { a.u[j] = p[j]; a.u[4 + j] = p[8 + j]; }
}
// B fragment: 8 contiguous dwords (fragment-major swizzled layout)
DEVFN void loadB(Frag& b, const unsigned* __restrict__ p) {
#pragma unroll
  for (int j = 0; j < 8; ++j) b.u[j] = p[j];
}

DEVFN unsigned pkh2(float a, float b) {
  union { _Float16 h[2]; unsigned u; } x;
  x.h[0] = (_Float16)a; x.h[1] = (_Float16)b; return x.u;
}
DEVFN float sigm(float x) { return 1.0f / (1.0f + __expf(-x)); }
DEVFN unsigned fenc(float f) {
  unsigned u = __float_as_uint(f);
  return (u & 0x80000000u) ? ~u : (u | 0x80000000u);
}
DEVFN float fdec(unsigned e) {
  unsigned u = (e & 0x80000000u) ? (e & 0x7FFFFFFFu) : ~e;
  return __uint_as_float(u);
}
DEVFN float vsel16(const v8h& a, const v8h& b, int c) {   // c is unrolled-const
  return (float)(c < 8 ? a[c] : b[c - 8]);
}

// ---------------------------------------------------------------------------
// Packing kernels
// ---------------------------------------------------------------------------

// inputs_all [65536][300] f32 -> pair dwords [65536][160] (K padded to 320)
__global__ void k_pack_x(const float* __restrict__ X, unsigned* __restrict__ out) {
  int idx = blockIdx.x * 256 + threadIdx.x;            // 65536*160
  if (idx >= 65536 * 160) return;
  int m = idx / 160, kp = idx % 160, k = kp * 2;
  float a = (k     < 300) ? X[(size_t)m * 300 + k]     : 0.0f;
  float b = (k + 1 < 300) ? X[(size_t)m * 300 + k + 1] : 0.0f;
  out[idx] = pkh2(a, b);
}

// W [Nsrc][Ksrc] f32 -> fragment-major swizzled B:
//   out[((kt*Ntiles + nt)*32 + lane)*8 + j] = pair (B[k][n], B[k+1][n]),
//   k = 2*(kt*16 + kpo(j,lane>>4)),  n = nt*16 + (lane&15),  B = W^T.
__global__ void k_pack_Bsw(const float* __restrict__ W, int Nsrc, int Ksrc,
                           int Ntiles, int Ktiles, unsigned* __restrict__ out) {
  int idx = blockIdx.x * 256 + threadIdx.x;
  if (idx >= Ktiles * Ntiles * 256) return;
  int j    = idx & 7;
  int lane = (idx >> 3) & 31;
  int rest = idx >> 8;
  int nt   = rest % Ntiles;
  int kt   = rest / Ntiles;
  int hi = lane >> 4, nl = lane & 15;
  int k  = 2 * (kt * 16 + kpo(j, hi));
  int n  = nt * 16 + nl;
  float a = (n < Nsrc && k     < Ksrc) ? W[(size_t)n * Ksrc + k]     : 0.0f;
  float b = (n < Nsrc && k + 1 < Ksrc) ? W[(size_t)n * Ksrc + k + 1] : 0.0f;
  out[idx] = pkh2(a, b);
}

__global__ void k_fill_u32(unsigned* __restrict__ p, unsigned v, int n) {
  int i = blockIdx.x * 256 + threadIdx.x;
  if (i < n) p[i] = v;
}

// ---------------------------------------------------------------------------
// WMMA GEMM, 4 N-tiles per wave: C_f16[M][N] = A_pk @ B_sw
// ---------------------------------------------------------------------------
__global__ void __launch_bounds__(256) k_gemm(
    const unsigned* __restrict__ Apk, int ldap,
    const unsigned* __restrict__ Bsw,
    _Float16* __restrict__ C,
    int Mtiles, int Ntiles, int Ktiles, int N) {
  int wid = blockIdx.x * 8 + (threadIdx.x >> 5);
  int Gt  = Ntiles >> 2;                       // N-tile groups of 4
  if (wid >= Mtiles * Gt) return;
  int mt = wid / Gt, nt0 = (wid % Gt) * 4;
  int lane = threadIdx.x & 31, n = lane & 15, hi = lane >> 4, m = lane & 15;
  const unsigned* arow = Apk + (size_t)(mt * 16 + m) * ldap + 4 * hi;
  const unsigned* brow = Bsw + ((size_t)nt0 * 32 + lane) * 8;
  v8f zero = {};
  v8f acc[4] = {zero, zero, zero, zero};
  unsigned bstride = (unsigned)Ntiles * 256;
  for (int kt = 0; kt < Ktiles; ++kt) {
    Frag a;
    loadA(a, arow + kt * 16);
#pragma unroll
    for (int t = 0; t < 4; ++t) {
      Frag b;
      loadB(b, brow + t * 256);
      acc[t] = wmma_f16(a, b, acc[t]);
    }
    brow += bstride;
  }
#pragma unroll
  for (int t = 0; t < 4; ++t)
#pragma unroll
    for (int r = 0; r < 8; ++r) {
      int row = mt * 16 + r + 8 * hi;
      C[(size_t)row * N + (nt0 + t) * 16 + n] = (_Float16)acc[t][r];
    }
}

// ---------------------------------------------------------------------------
// Word-level GRU recurrence.  32 blocks: (16 sentence groups) x (2 dirs).
// Per step: GH = h @ Whh^T (h in LDS, A-frag hoisted over 6 N-tiles), gates.
// ---------------------------------------------------------------------------
__global__ void __launch_bounds__(256) k_word_gru(
    const unsigned* __restrict__ Bh,      // Whh^T swizzled (Ktiles=8,Ntiles=48)
    const _Float16* __restrict__ GI,      // [65536][768]
    const float* __restrict__ bih, const float* __restrict__ bhh,
    unsigned* __restrict__ hs) {          // h_states pairs [256][512][128]
  __shared__ unsigned hbuf[16][128];      // h (16 sentences x 256 f16)
  __shared__ float    gh[16][768];
  __shared__ float    bi[768], bh2[768];
  const int tid = threadIdx.x;
  const int dir = blockIdx.x & 1;
  const int ns0 = (blockIdx.x >> 1) * 16;
  for (int i = tid; i < 16 * 128; i += 256) ((unsigned*)hbuf)[i] = 0u;
  for (int i = tid; i < 768; i += 256) { bi[i] = bih[i]; bh2[i] = bhh[i]; }
  __syncthreads();
  const int lane = tid & 31, wv = tid >> 5, n = lane & 15, hi = lane >> 4, m = lane & 15;
  const int grow = tid >> 4;              // gate-phase row 0..15
  const int cb   = (tid & 15) * 16;       // gate-phase col base
  const unsigned* bwave = Bh + ((size_t)(wv * 6) * 32 + lane) * 8;
  for (int s = 0; s < 256; ++s) {
    int t = dir ? (255 - s) : s;
    // --- GH = h @ Whh^T : one A-frag per k-tile reused over 6 N-tiles ---
    {
      v8f zero = {};
      v8f acc[6] = {zero, zero, zero, zero, zero, zero};
      const unsigned* bp = bwave;
#pragma unroll
      for (int kt = 0; kt < 8; ++kt) {
        Frag a;
        loadA(a, &hbuf[m][kt * 16 + 4 * hi]);
#pragma unroll
        for (int q = 0; q < 6; ++q) {
          Frag b;
          loadB(b, bp + q * 256);
          acc[q] = wmma_f16(a, b, acc[q]);
        }
        bp += 48 * 256;
      }
#pragma unroll
      for (int q = 0; q < 6; ++q)
#pragma unroll
        for (int r = 0; r < 8; ++r)
          gh[r + 8 * hi][(wv * 6 + q) * 16 + n] = acc[q][r];
    }
    __syncthreads();
    // --- gates: r=sig(gi_r+gh_r), z=sig(gi_z+gh_z), n=tanh(gi_n + r*gh_n) ---
    const _Float16* gi = GI + ((size_t)(ns0 + grow) * 256 + t) * 768;
    v8h gr0 = *(const v8h*)(gi + cb),       gr1 = *(const v8h*)(gi + cb + 8);
    v8h gz0 = *(const v8h*)(gi + 256 + cb), gz1 = *(const v8h*)(gi + 256 + cb + 8);
    v8h gn0 = *(const v8h*)(gi + 512 + cb), gn1 = *(const v8h*)(gi + 512 + cb + 8);
    int pos = dir ? (256 + t) : t;
    unsigned* hsrow = hs + ((size_t)(ns0 + grow) * 512 + pos) * 128;
#pragma unroll
    for (int p = 0; p < 8; ++p) {
      union { unsigned u; _Float16 h[2]; } ho;
      ho.u = hbuf[grow][cb / 2 + p];
      float hn2[2];
#pragma unroll
      for (int e = 0; e < 2; ++e) {
        int c = 2 * p + e, col = cb + c;
        float rg = sigm(vsel16(gr0, gr1, c) + bi[col] + gh[grow][col] + bh2[col]);
        float zg = sigm(vsel16(gz0, gz1, c) + bi[256 + col] +
                        gh[grow][256 + col] + bh2[256 + col]);
        float ng = tanhf(vsel16(gn0, gn1, c) + bi[512 + col] +
                         rg * (gh[grow][512 + col] + bh2[512 + col]));
        hn2[e] = (1.0f - zg) * ng + zg * (float)ho.h[e];
      }
      unsigned pk = pkh2(hn2[0], hn2[1]);
      hbuf[grow][cb / 2 + p] = pk;
      hsrow[cb / 2 + p] = pk;
    }
    __syncthreads();
  }
}

// ---------------------------------------------------------------------------
// Convolutions: implicit-im2col WMMA, one wave per (ns, kw, pos-tile) with
// all 7 filter tiles register-resident.  256*6*32 = 49152 waves.
// ---------------------------------------------------------------------------
__global__ void __launch_bounds__(256) k_conv(
    const unsigned* __restrict__ hs,    // [256][512][128] pairs
    const unsigned* __restrict__ Bc,    // swizzled conv weights (all widths)
    unsigned* __restrict__ fmax) {      // [256][600] encoded running max
  const int wid = blockIdx.x * 8 + (threadIdx.x >> 5);
  int ns  = wid / (6 * 32);
  int r0  = wid % (6 * 32);
  int kwi = r0 / 32;
  int pt  = r0 % 32;
  int kw  = kwi + 1;
  int pad = (kw - 1) >> 1;
  int outlen = 512 + 2 * pad - kw + 1;
  const int lane = threadIdx.x & 31, n = lane & 15, hi = lane >> 4, m = lane & 15;
  const int coffs[6] = {0, 14336, 43008, 86016, 143360, 215040};
  const unsigned* B = Bc + coffs[kwi] + (size_t)lane * 8;
  const unsigned* H = hs + (size_t)ns * 512 * 128;
  int p_m = pt * 16 + m;                // im2col row this lane supplies
  v8f zero = {};
  v8f acc[7] = {zero, zero, zero, zero, zero, zero, zero};
  int ktiles = kw * 8;                  // K = kw*256
  for (int kt = 0; kt < ktiles; ++kt) {
    int dk   = kt >> 3;
    int rowh = p_m - pad + dk;
    int base = (kt & 7) * 16;
    Frag a;
    if (rowh >= 0 && rowh < 512) {
      loadA(a, H + (size_t)rowh * 128 + base + 4 * hi);
    } else {
#pragma unroll
      for (int j = 0; j < 8; ++j) a.u[j] = 0u;
    }
    const unsigned* bp = B + (size_t)(kt * 7) * 256;
#pragma unroll
    for (int nt = 0; nt < 7; ++nt) {
      Frag b;
      loadB(b, bp + nt * 256);
      acc[nt] = wmma_f16(a, b, acc[nt]);
    }
  }
  // global-max over positions, then fold lane-halves, then atomicMax
#pragma unroll
  for (int nt = 0; nt < 7; ++nt) {
    float mx = -3.0e38f;
#pragma unroll
    for (int r = 0; r < 8; ++r) {
      int p = pt * 16 + r + 8 * hi;
      if (p < outlen) mx = fmaxf(mx, acc[nt][r]);
    }
    mx = fmaxf(mx, __shfl_xor(mx, 16, 32));
    int o = nt * 16 + n;
    if (hi == 0 && o < 100)
      atomicMax(&fmax[(size_t)ns * 600 + kwi * 100 + o], fenc(mx));
  }
}

// decode max, add conv bias, sigmoid -> feats, pair-packed f16 [256][304]
__global__ void k_feats(const unsigned* __restrict__ fmax,
                        const float* cb1, const float* cb2, const float* cb3,
                        const float* cb4, const float* cb5, const float* cb6,
                        unsigned* __restrict__ fpk) {
  int idx = blockIdx.x * 256 + threadIdx.x;            // 256*304
  if (idx >= 256 * 304) return;
  int ns = idx / 304, kp = idx % 304;
  const float* cbs[6] = {cb1, cb2, cb3, cb4, cb5, cb6};
  float v[2];
#pragma unroll
  for (int e = 0; e < 2; ++e) {
    int k = kp * 2 + e;
    float f = 0.0f;
    if (k < 600) {
      int kwi = k / 100, o = k % 100;
      f = sigm(fdec(fmax[(size_t)ns * 600 + k]) + cbs[kwi][o]);
    }
    v[e] = f;
  }
  fpk[idx] = pkh2(v[0], v[1]);
}

// ---------------------------------------------------------------------------
// Sentence-level GRU (batch 1), 2 blocks (fwd/bwd).  Row 0 of the tile
// carries the state; rows 1..15 stay zero.  Accumulates the state mean.
// ---------------------------------------------------------------------------
__global__ void __launch_bounds__(256) k_sent_gru(
    const unsigned* __restrict__ Bsh,   // Whh_s^T swizzled (Ktiles=8,Ntiles=48)
    const _Float16* __restrict__ GIs,   // [256][768]
    const float* __restrict__ bih, const float* __restrict__ bhh,
    float* __restrict__ meanout) {      // [2][256]
  __shared__ unsigned hbuf[16][128];
  __shared__ float ghs[768];
  __shared__ float hsum[256];
  int tid = threadIdx.x, dir = blockIdx.x;
  for (int i = tid; i < 16 * 128; i += 256) ((unsigned*)hbuf)[i] = 0u;
  hsum[tid] = 0.0f;
  __syncthreads();
  int lane = tid & 31, wv = tid >> 5, n = lane & 15, hi = lane >> 4, m = lane & 15;
  const unsigned* bwave = Bsh + ((size_t)(wv * 6) * 32 + lane) * 8;
  for (int s = 0; s < 256; ++s) {
    int t = dir ? (255 - s) : s;
    {
      v8f zero = {};
      v8f acc[6] = {zero, zero, zero, zero, zero, zero};
      const unsigned* bp = bwave;
#pragma unroll
      for (int kt = 0; kt < 8; ++kt) {
        Frag a;
        loadA(a, &hbuf[m][kt * 16 + 4 * hi]);
#pragma unroll
        for (int q = 0; q < 6; ++q) {
          Frag b;
          loadB(b, bp + q * 256);
          acc[q] = wmma_f16(a, b, acc[q]);
        }
        bp += 48 * 256;
      }
      if (hi == 0)
#pragma unroll
        for (int q = 0; q < 6; ++q) ghs[(wv * 6 + q) * 16 + n] = acc[q][0];
    }
    __syncthreads();
    if (tid < 128) {
      const _Float16* gi = GIs + (size_t)t * 768;
      union { unsigned u; _Float16 h[2]; } ho;
      ho.u = hbuf[0][tid];
      float hn2[2];
#pragma unroll
      for (int e = 0; e < 2; ++e) {
        int col = tid * 2 + e;
        float rg = sigm((float)gi[col] + bih[col] + ghs[col] + bhh[col]);
        float zg = sigm((float)gi[256 + col] + bih[256 + col] +
                        ghs[256 + col] + bhh[256 + col]);
        float ng = tanhf((float)gi[512 + col] + bih[512 + col] +
                         rg * (ghs[512 + col] + bhh[512 + col]));
        hn2[e] = (1.0f - zg) * ng + zg * (float)ho.h[e];
        hsum[col] += hn2[e];
      }
      hbuf[0][tid] = pkh2(hn2[0], hn2[1]);
    }
    __syncthreads();
  }
  meanout[dir * 256 + tid] = hsum[tid] * (1.0f / 256.0f);
}

// ---------------------------------------------------------------------------
// Final FC head (tiny, one block, f32 VALU)
// ---------------------------------------------------------------------------
__global__ void k_head(const float* __restrict__ meanbuf,
                       const float* fc1w, const float* fc1b,
                       const float* fc2w, const float* fc2b,
                       const float* fc3w, const float* fc3b,
                       float* __restrict__ out) {
  __shared__ float hdoc[256];
  __shared__ float x1[128];
  __shared__ float x2[32];
  int tid = threadIdx.x;                 // 256 threads
  hdoc[tid] = 0.5f * (meanbuf[tid] + meanbuf[256 + tid]);
  __syncthreads();
  if (tid < 128) {
    float s = fc1b[tid];
    for (int j = 0; j < 256; ++j) s += fc1w[tid * 256 + j] * hdoc[j];
    x1[tid] = sigm(s);
  }
  __syncthreads();
  if (tid < 32) {
    float s = fc2b[tid];
    for (int j = 0; j < 128; ++j) s += fc2w[tid * 128 + j] * x1[j];
    x2[tid] = sigm(s);
  }
  __syncthreads();
  if (tid == 0) {
    float s = fc3b[0];
    for (int j = 0; j < 32; ++j) s += fc3w[j] * x2[j];
    out[0] = sigm(s);
  }
}

// ---------------------------------------------------------------------------
// Host launch
// ---------------------------------------------------------------------------
extern "C" void kernel_launch(void* const* d_in, const int* in_sizes, int n_in,
                              void* d_out, int out_size, void* d_ws, size_t ws_size,
                              hipStream_t stream) {
  const float* inputs = (const float*)d_in[0];
  const float* wih_w  = (const float*)d_in[1];
  const float* whh_w  = (const float*)d_in[2];
  const float* bih_w  = (const float*)d_in[3];
  const float* bhh_w  = (const float*)d_in[4];
  const float* wih_s  = (const float*)d_in[5];
  const float* whh_s  = (const float*)d_in[6];
  const float* bih_s  = (const float*)d_in[7];
  const float* bhh_s  = (const float*)d_in[8];
  const float* cw[6]; const float* cbv[6];
  for (int i = 0; i < 6; ++i) { cw[i] = (const float*)d_in[9 + 2 * i];
                                cbv[i] = (const float*)d_in[10 + 2 * i]; }
  const float* fc1w = (const float*)d_in[21]; const float* fc1b = (const float*)d_in[22];
  const float* fc2w = (const float*)d_in[23]; const float* fc2b = (const float*)d_in[24];
  const float* fc3w = (const float*)d_in[25]; const float* fc3b = (const float*)d_in[26];
  float* out = (float*)d_out;

  // ---- workspace carve-up (bytes, 256-aligned) ----
  char* ws = (char*)d_ws;
  size_t off = 0;
  auto carve = [&](size_t bytes) {
    size_t r = off; off = (off + bytes + 255) & ~(size_t)255; return r;
  };
  unsigned* x16  = (unsigned*)(ws + carve((size_t)65536 * 160 * 4)); // 41.9MB
  unsigned* Bw   = (unsigned*)(ws + carve((size_t)160 * 768 * 4));
  unsigned* Bh   = (unsigned*)(ws + carve((size_t)128 * 768 * 4));
  unsigned* Bsi  = (unsigned*)(ws + carve((size_t)304 * 768 * 4));
  unsigned* Bsh  = (unsigned*)(ws + carve((size_t)128 * 768 * 4));
  unsigned* Bc   = (unsigned*)(ws + carve((size_t)301056 * 4));      // 1.2MB
  _Float16* GI   = (_Float16*)(ws + carve((size_t)65536 * 768 * 2)); // 100.7MB
  unsigned* hs   = (unsigned*)(ws + carve((size_t)256 * 512 * 128 * 4)); // 67.1MB
  unsigned* fmax = (unsigned*)(ws + carve((size_t)256 * 600 * 4));
  unsigned* fpk  = (unsigned*)(ws + carve((size_t)256 * 304 * 4));
  _Float16* GIs  = (_Float16*)(ws + carve((size_t)256 * 768 * 2));
  float*    mean = (float*)   (ws + carve((size_t)512 * 4));
  (void)ws_size; (void)n_in; (void)in_sizes; (void)out_size;

  // ---- pack inputs and weights into WMMA-fragment layouts ----
  k_pack_x<<<40960, 256, 0, stream>>>(inputs, x16);
  k_pack_Bsw<<<480, 256, 0, stream>>>(wih_w, 768, 300, 48, 10, Bw);
  k_pack_Bsw<<<384, 256, 0, stream>>>(whh_w, 768, 256, 48,  8, Bh);
  k_pack_Bsw<<<912, 256, 0, stream>>>(wih_s, 768, 600, 48, 19, Bsi);
  k_pack_Bsw<<<384, 256, 0, stream>>>(whh_s, 768, 256, 48,  8, Bsh);
  const int coffs[6] = {0, 14336, 43008, 86016, 143360, 215040};
  for (int k = 1; k <= 6; ++k) {
    int Ktiles = k * 8;                               // K = k*256, Ntiles = 7
    k_pack_Bsw<<<(Ktiles * 7 * 256 + 255) / 256, 256, 0, stream>>>(
        cw[k - 1], 100, k * 256, 7, Ktiles, Bc + coffs[k - 1]);
  }

  // ---- GI = X @ Wih^T  (M=65536, N=768, K=320) ----
  k_gemm<<<6144, 256, 0, stream>>>(x16, 160, Bw, GI, 4096, 48, 10, 768);

  // ---- word-level bi-GRU recurrence (16 sentence groups x 2 dirs) ----
  k_word_gru<<<32, 256, 0, stream>>>(Bh, GI, bih_w, bhh_w, hs);

  // ---- convs + global max (init running max first) ----
  k_fill_u32<<<600, 256, 0, stream>>>(fmax, 0u, 256 * 600);
  k_conv<<<6144, 256, 0, stream>>>(hs, Bc, fmax);
  k_feats<<<304, 256, 0, stream>>>(fmax, cbv[0], cbv[1], cbv[2], cbv[3], cbv[4],
                                   cbv[5], fpk);

  // ---- GIs = feats @ Wih_s^T (M=256, N=768, K=608) ----
  k_gemm<<<24, 256, 0, stream>>>(fpk, 304, Bsi, GIs, 16, 48, 19, 768);

  // ---- sentence-level bi-GRU + means ----
  k_sent_gru<<<2, 256, 0, stream>>>(Bsh, GIs, bih_s, bhh_s, mean);

  // ---- FC head ----
  k_head<<<1, 256, 0, stream>>>(mean, fc1w, fc1b, fc2w, fc2b, fc3w, fc3b, out);
}
